// graph_encoder_68856915690120
// MI455X (gfx1250) — compile-verified, matrix operand order checked by reading, not run on hardware
//
#include <hip/hip_runtime.h>

#define NNODES 10000
#define KDIM   10000
#define KPAD   10048   // KDIM padded to a multiple of 64 for the double-wide WMMA K loop
#define H1     256
#define H2     128
#define NEDGE  320000

typedef unsigned short ushort_t;
typedef __attribute__((ext_vector_type(16))) __bf16   bf16x16;
typedef __attribute__((ext_vector_type(8)))  float    f32x8;
typedef __attribute__((ext_vector_type(4)))  float    f32x4;
typedef __attribute__((ext_vector_type(4)))  unsigned u32x4;
typedef __attribute__((ext_vector_type(2)))  unsigned u32x2;

union FragBF {
    bf16x16 v;
    u32x4   q[2];
};

__device__ __forceinline__ ushort_t f2bf(float f) {
    unsigned u = __builtin_bit_cast(unsigned, f);
    u += 0x7FFFu + ((u >> 16) & 1u);      // round-to-nearest-even
    return (ushort_t)(u >> 16);
}

// Load a 16x32 bf16 A/B fragment from a row-major [rows][ldShorts] bf16 buffer.
// Matches CDNA5 16-bit A layout: lane<16 -> K {k..k+7, k+16..k+23},
// lane>=16 -> K {k+8..k+15, k+24..k+31}, row = row0 + (lane&15).
__device__ __forceinline__ bf16x16 load_frag_global(const ushort_t* __restrict__ base,
                                                    int ldShorts, int row0, int k) {
    const int lane = threadIdx.x & 31;
    const int r    = row0 + (lane & 15);
    const int ko   = k + ((lane >> 4) << 3);
    const u32x4* p = (const u32x4*)(base + (size_t)r * ldShorts + ko);
    FragBF f;
    f.q[0] = p[0];   // K = ko .. ko+7
    f.q[1] = p[2];   // K = ko+16 .. ko+23
    return f.v;
}

// Same fragment pattern from an LDS tile: pitch in shorts, ksub = 0 or 32.
__device__ __forceinline__ bf16x16 load_frag_lds(const ushort_t* As, int pitch, int ksub) {
    const int lane = threadIdx.x & 31;
    const int r    = lane & 15;
    const int ko   = ksub + ((lane >> 4) << 3);
    const u32x4* p = (const u32x4*)(As + r * pitch + ko);
    FragBF f;
    f.q[0] = p[0];
    f.q[1] = p[2];
    return f.v;
}

// Store a 16x16 f32 C tile (8 VGPRs/lane) per documented C/D layout.
__device__ __forceinline__ void store_tile_f32(float* __restrict__ out, size_t ld,
                                               int m0, int n0, f32x8 c) {
    const int lane = threadIdx.x & 31;
    const int n    = n0 + (lane & 15);
    const int mb   = m0 + ((lane >> 4) << 3);
#pragma unroll
    for (int j = 0; j < 8; ++j)
        out[(size_t)(mb + j) * ld + n] = c[j];
}

// Stage 4 consecutive feat f32 values -> 4 bf16 into LDS as one 8-byte store.
// Wave-uniform fast path when the whole 64-wide K chunk is in range.
__device__ __forceinline__ void stage_tile(ushort_t* __restrict__ dstLds,
                                           const float* __restrict__ arow,
                                           int kk, int ac) {
    f32x4 v = {0.0f, 0.0f, 0.0f, 0.0f};
    if (kk + 64 <= KDIM) {                      // uniform: no per-lane EXEC dance
        v = *(const f32x4*)(arow + kk + ac);
    } else if (kk + ac + 4 <= KDIM) {           // tail chunk only (last iteration)
        v = *(const f32x4*)(arow + kk + ac);
    }
    u32x2 packed;
    packed.x = (unsigned)f2bf(v.x) | ((unsigned)f2bf(v.y) << 16);
    packed.y = (unsigned)f2bf(v.z) | ((unsigned)f2bf(v.w) << 16);
    *(u32x2*)dstLds = packed;                   // 8B-aligned: pitch 144B, ac*2 % 8 == 0
}

#define WMMA_BF16(A, B, C) \
    __builtin_amdgcn_wmma_f32_16x16x32_bf16(false, (A), false, (B), (short)0, (C), false, false)

// ---------------------------------------------------------------- utilities
__global__ void zero_kernel(float* __restrict__ p, int n) {
    for (int i = blockIdx.x * blockDim.x + threadIdx.x; i < n; i += gridDim.x * blockDim.x)
        p[i] = 0.0f;
}

__global__ void deg_kernel(const int* __restrict__ dst, float* __restrict__ deg) {
    int e = blockIdx.x * blockDim.x + threadIdx.x;
    if (e < NEDGE)
        __hip_atomic_fetch_add(&deg[dst[e]], 1.0f, __ATOMIC_RELAXED, __HIP_MEMORY_SCOPE_AGENT);
}

__global__ void inv_kernel(const float* __restrict__ deg, float* __restrict__ inv) {
    int i = blockIdx.x * blockDim.x + threadIdx.x;
    if (i < NNODES) inv[i] = 1.0f / (deg[i] + 1.0f);
}

// Convert a row-major f32 [rows][cols] weight to bf16 [rows][colsPad], zero-padded.
__global__ void cvtw_kernel(const float* __restrict__ w, ushort_t* __restrict__ out,
                            int rows, int cols, int colsPad) {
    int idx = blockIdx.x * blockDim.x + threadIdx.x;
    int total = rows * colsPad;
    if (idx >= total) return;
    int r = idx / colsPad, c = idx - r * colsPad;
    out[idx] = (c < cols) ? f2bf(w[(size_t)r * cols + c]) : (ushort_t)0;
}

// --------------------------------------------------------------- GEMM 1
// ht1[10000,256] = feat[10000,10000] @ W1^T.
// 16x64 double-buffered LDS tile, f32->bf16 on the fly, 4 WMMAs per barrier.
__global__ __launch_bounds__(256)
void gemm1_kernel(const float* __restrict__ feat, const ushort_t* __restrict__ w1bf,
                  float* __restrict__ ht1) {
    __shared__ ushort_t As[2][16 * 72];        // pitch 72 shorts (144B): aligned + conflict-free
    const int m0   = blockIdx.x * 16;
    const int t    = threadIdx.x;
    const int wave = t >> 5;
    const int ar   = t >> 4;                   // 0..15 : tile row this thread fills
    const int ac   = (t & 15) * 4;             // 0..60 : k-quad this thread fills
    const int nb0  = wave * 32;                // each wave owns 32 output columns

    f32x8 acc0 = {}; f32x8 acc1 = {};
    const float*    arow  = feat + (size_t)(m0 + ar) * KDIM;
    ushort_t* const myLds0 = &As[0][ar * 72 + ac];
    ushort_t* const myLds1 = &As[1][ar * 72 + ac];

    stage_tile(myLds0, arow, 0, ac);
    __syncthreads();

    int buf = 0;
    for (int k = 0; k < KPAD; k += 64) {
        const int kn = k + 64;
        if (kn < KPAD)                          // stage next tile into the other buffer
            stage_tile(buf ? myLds0 : myLds1, arow, kn, ac);
        __builtin_prefetch(arow + k + 192, 0, 1);   // global_prefetch_b8 two tiles ahead

        const ushort_t* cur = As[buf];
        bf16x16 a0  = load_frag_lds(cur, 72, 0);
        bf16x16 a1  = load_frag_lds(cur, 72, 32);
        bf16x16 b00 = load_frag_global(w1bf, KPAD, nb0,      k);
        bf16x16 b01 = load_frag_global(w1bf, KPAD, nb0 + 16, k);
        bf16x16 b10 = load_frag_global(w1bf, KPAD, nb0,      k + 32);
        bf16x16 b11 = load_frag_global(w1bf, KPAD, nb0 + 16, k + 32);
        acc0 = WMMA_BF16(a0, b00, acc0);
        acc1 = WMMA_BF16(a0, b01, acc1);
        acc0 = WMMA_BF16(a1, b10, acc0);
        acc1 = WMMA_BF16(a1, b11, acc1);

        __syncthreads();                        // next-buffer writes done AND cur reads done
        buf ^= 1;
    }
    store_tile_f32(ht1, H1, m0, nb0,      acc0);
    store_tile_f32(ht1, H1, m0, nb0 + 16, acc1);
}

// --------------------------------------------------------------- scatter
// agg[dst] += ew * ht[src]   (one block per edge, one thread per feature column)
__global__ void scatter_kernel(const float* __restrict__ ht, const int* __restrict__ src,
                               const int* __restrict__ dst, const float* __restrict__ ew,
                               float* __restrict__ agg, int cols) {
    const int e = blockIdx.x;
    const int c = threadIdx.x;
    const int s = src[e], d = dst[e];
    const float w = ew[e];
    __hip_atomic_fetch_add(&agg[(size_t)d * cols + c], w * ht[(size_t)s * cols + c],
                           __ATOMIC_RELAXED, __HIP_MEMORY_SCOPE_AGENT);
}

// h = relu((agg + ht1) * inv + b1)  ->  bf16
__global__ void fin1_kernel(const float* __restrict__ agg, const float* __restrict__ ht1,
                            const float* __restrict__ inv, const float* __restrict__ b1,
                            ushort_t* __restrict__ hbf) {
    int idx = blockIdx.x * blockDim.x + threadIdx.x;
    if (idx >= NNODES * H1) return;
    int i = idx >> 8, c = idx & 255;
    float v = (agg[idx] + ht1[idx]) * inv[i] + b1[c];
    hbf[idx] = f2bf(fmaxf(v, 0.0f));
}

// --------------------------------------------------------------- GEMM 2
// ht2[10000,128] = h[10000,256] @ W2^T  (K=256, fully unrolled, L2-resident operands)
__global__ __launch_bounds__(256)
void gemm2_kernel(const ushort_t* __restrict__ hbf, const ushort_t* __restrict__ w2bf,
                  float* __restrict__ ht2) {
    const int m0 = blockIdx.x * 16;
    const int n0 = (threadIdx.x >> 5) * 16;
    f32x8 acc = {};
#pragma unroll
    for (int k = 0; k < H1; k += 32) {
        bf16x16 a = load_frag_global(hbf,  H1, m0, k);
        bf16x16 b = load_frag_global(w2bf, H1, n0, k);
        acc = WMMA_BF16(a, b, acc);
    }
    store_tile_f32(ht2, H2, m0, n0, acc);
}

// z = (agg + ht2) * inv + b2  -> d_out (f32) and bf16 copy for the adj GEMM
__global__ void fin2_kernel(const float* __restrict__ agg, const float* __restrict__ ht2,
                            const float* __restrict__ inv, const float* __restrict__ b2,
                            float* __restrict__ zout, ushort_t* __restrict__ zbf) {
    int idx = blockIdx.x * blockDim.x + threadIdx.x;
    if (idx >= NNODES * H2) return;
    int i = idx >> 7, c = idx & 127;
    float v = (agg[idx] + ht2[idx]) * inv[i] + b2[c];
    zout[idx] = v;
    zbf[idx]  = f2bf(v);
}

// --------------------------------------------------------------- GEMM 3
// adj[10000,10000] = z @ z^T   (K=128; z bf16 is 2.5MB -> fully L2-resident; store bound)
__global__ __launch_bounds__(256)
void gemm3_kernel(const ushort_t* __restrict__ zbf, float* __restrict__ adj) {
    const int m0 = blockIdx.x * 16;
    const int nt = blockIdx.y * 8 + (threadIdx.x >> 5);
    if (nt >= NNODES / 16) return;             // wave-uniform: EXEC stays all-ones
    const int n0 = nt * 16;
    f32x8 acc = {};
#pragma unroll
    for (int k = 0; k < H2; k += 32) {
        bf16x16 a = load_frag_global(zbf, H2, m0, k);
        bf16x16 b = load_frag_global(zbf, H2, n0, k);   // columns of z^T == rows of z
        acc = WMMA_BF16(a, b, acc);
    }
    store_tile_f32(adj, NNODES, m0, n0, acc);
}

// ---------------------------------------------------------------- launch
extern "C" void kernel_launch(void* const* d_in, const int* in_sizes, int n_in,
                              void* d_out, int out_size, void* d_ws, size_t ws_size,
                              hipStream_t stream) {
    const float* feat = (const float*)d_in[0];
    const float* W1   = (const float*)d_in[1];
    const float* b1   = (const float*)d_in[2];
    const float* W2   = (const float*)d_in[3];
    const float* b2   = (const float*)d_in[4];
    const float* ew   = (const float*)d_in[5];
    const int*   src  = (const int*)d_in[6];
    const int*   dst  = (const int*)d_in[7];

    float* out_z   = (float*)d_out;
    float* out_adj = out_z + (size_t)NNODES * H2;

    char* p = (char*)d_ws;
    auto alloc = [&](size_t bytes) -> char* {
        char* q = p;
        p += (bytes + 255) & ~(size_t)255;
        return q;
    };
    float*    deg  = (float*)   alloc((size_t)NNODES * 4);
    float*    inv  = (float*)   alloc((size_t)NNODES * 4);
    ushort_t* w1bf = (ushort_t*)alloc((size_t)H1 * KPAD * 2);
    ushort_t* w2bf = (ushort_t*)alloc((size_t)H2 * H1 * 2);
    float*    ht1  = (float*)   alloc((size_t)NNODES * H1 * 4);
    float*    agg  = (float*)   alloc((size_t)NNODES * H1 * 4);
    ushort_t* hbf  = (ushort_t*)alloc((size_t)NNODES * H1 * 2);
    float*    ht2  = (float*)   alloc((size_t)NNODES * H2 * 4);
    ushort_t* zbf  = (ushort_t*)alloc((size_t)NNODES * H2 * 2);

    // degrees and normalization
    zero_kernel<<<64, 256, 0, stream>>>(deg, NNODES);
    deg_kernel<<<(NEDGE + 255) / 256, 256, 0, stream>>>(dst, deg);
    inv_kernel<<<(NNODES + 255) / 256, 256, 0, stream>>>(deg, inv);

    // weight conversion (bf16, K-padded)
    cvtw_kernel<<<(H1 * KPAD + 255) / 256, 256, 0, stream>>>(W1, w1bf, H1, KDIM, KPAD);
    cvtw_kernel<<<(H2 * H1 + 255) / 256, 256, 0, stream>>>(W2, w2bf, H2, H1, H1);

    // layer 1
    gemm1_kernel<<<NNODES / 16, 256, 0, stream>>>(feat, w1bf, ht1);
    zero_kernel<<<2048, 256, 0, stream>>>(agg, NNODES * H1);
    scatter_kernel<<<NEDGE, H1, 0, stream>>>(ht1, src, dst, ew, agg, H1);
    fin1_kernel<<<(NNODES * H1 + 255) / 256, 256, 0, stream>>>(agg, ht1, inv, b1, hbf);

    // layer 2
    gemm2_kernel<<<NNODES / 16, 256, 0, stream>>>(hbf, w2bf, ht2);
    zero_kernel<<<2048, 256, 0, stream>>>(agg, NNODES * H2);
    scatter_kernel<<<NEDGE, H2, 0, stream>>>(ht2, src, dst, ew, agg, H2);
    fin2_kernel<<<(NNODES * H2 + 255) / 256, 256, 0, stream>>>(agg, ht2, inv, b2, out_z, zbf);

    // dense reconstruction
    gemm3_kernel<<<dim3(NNODES / 16, (NNODES / 16 + 7) / 8), 256, 0, stream>>>(zbf, out_adj);
}
